// WeightOnlyLinear_19765439496679
// MI455X (gfx1250) — matrix operand order repaired and checked by reading, not compiled, not run on hardware
//
#include <hip/hip_runtime.h>

#define BATCH 64
#define IN_F 8192
#define OUT_F 8192
#define GROUPSIZE 128
#define NGROUPS (IN_F / GROUPSIZE)           // 64
#define KSPLIT 8
#define GROUPS_PER_SPLIT (NGROUPS / KSPLIT)  // 8
#define NT 2                                  // N-tiles per wave (share A regs)

typedef _Float16 v16h __attribute__((ext_vector_type(16)));
typedef _Float16 h8   __attribute__((ext_vector_type(8)));
typedef _Float16 h2   __attribute__((ext_vector_type(2)));
typedef float    v8f  __attribute__((ext_vector_type(8)));

union HF {
    v16h v;
    h8   q[2];
    h2   h[8];
};

__device__ __forceinline__ h2 make_h2(float a, float b) {
    h2 r; r.x = (_Float16)a; r.y = (_Float16)b; return r;
}

// qs pre-shifted so the wanted nibbles sit at bits [3:0] and [19:16].
// Magic-exponent expand to (1024+w0, 1024+w1), then exact f16 dequant:
// (w - zp) * scale via v_and_or + v_pk_add + v_pk_mul (ints <=1039 exact).
__device__ __forceinline__ h2 dq_pair(unsigned qs, h2 s2, h2 cz) {
    unsigned t = (qs & 0x000F000Fu) | 0x64006400u;
    h2 w = __builtin_bit_cast(h2, t);
    return (w - cz) * s2;
}

// ---------------------------------------------------------------------------
// Pass 1: x fp32 -> f16 with K-relabeling sigma=[0,4,1,5,2,6,3,7] inside every
// 8-K block, so GEMM A loads are plain 16B loads matching the B nibble order.
// ---------------------------------------------------------------------------
__global__ __launch_bounds__(256)
void cvt_x_kernel(const float* __restrict__ x, _Float16* __restrict__ xh)
{
    const int t = blockIdx.x * blockDim.x + threadIdx.x;  // one 8-K block
    const float4 a = ((const float4*)(x + (size_t)t * 8))[0];
    const float4 b = ((const float4*)(x + (size_t)t * 8))[1];
    HF o;
    o.h[0] = make_h2(a.x, b.x);
    o.h[1] = make_h2(a.y, b.y);
    o.h[2] = make_h2(a.z, b.z);
    o.h[3] = make_h2(a.w, b.w);
    *(h8*)(xh + (size_t)t * 8) = o.q[0];
}

// ---------------------------------------------------------------------------
// Pass 2: fused W4 dequant + GEMM. One wave: 2 N-tiles x 4 M-tiles x (1/8 K).
// A fragments loaded once per K-step feed 8 WMMAs; two B buffers (one per
// N-tile) space out WMMA->VALU register reuse. K-split partials merged with
// LDS float atomics (ds_add_f32).
// ---------------------------------------------------------------------------
__global__ __launch_bounds__(256)
void w4a16_wmma_kernel(const _Float16* __restrict__ xh,
                       const int*      __restrict__ qweight,
                       const int*      __restrict__ qzeros,
                       const float*    __restrict__ scales,
                       const float*    __restrict__ bias,
                       float*          __restrict__ out)
{
    __shared__ float acc[NT][4][256];               // 8 KB accumulation image

    const int lane = threadIdx.x & 31;
    const int ks   = threadIdx.x >> 5;              // K split id = wave id
    const int hi   = lane >> 4;
    const int l16  = lane & 15;

    const int nTile0 = blockIdx.x * NT;             // block owns an N-tile pair
    const int n0 = nTile0 * 16 + l16;               // lane's column, N-tile 0
    const int n1 = n0 + 16;                         // lane's column, N-tile 1

    const int*      qcol0 = qweight + n0;           // row stride = OUT_F
    const int*      qcol1 = qweight + n1;
    const _Float16* xbase = xh + 16 * hi;

    v8f c[NT][4] = {{{}, {}, {}, {}}, {{}, {}, {}, {}}};

    for (int g = 0; g < GROUPS_PER_SPLIT; ++g) {
        const int ga = ks * GROUPS_PER_SPLIT + g;
        // per-(group, column) constants for both N-tiles
        const int   zq0 = qzeros[ga * (OUT_F / 8) + (n0 >> 3)];
        const int   zq1 = qzeros[ga * (OUT_F / 8) + (n1 >> 3)];
        const unsigned zp0 = (unsigned)((((zq0 >> (4 * (n0 & 7))) & 15) + 1) & 15);
        const unsigned zp1 = (unsigned)((((zq1 >> (4 * (n1 & 7))) & 15) + 1) & 15);
        // (1024+zp) as f16 is exactly 0x6400|zp: build packed const integer-only
        const h2 cz0 = __builtin_bit_cast(h2, 0x64006400u | zp0 * 0x10001u);
        const h2 cz1 = __builtin_bit_cast(h2, 0x64006400u | zp1 * 0x10001u);
        const float sc0 = scales[ga * OUT_F + n0];
        const float sc1 = scales[ga * OUT_F + n1];
        const h2 s20 = make_h2(sc0, sc0);
        const h2 s21 = make_h2(sc1, sc1);
        const int rbase = ga * 16 + 2 * hi;

#pragma unroll
        for (int s = 0; s < 4; ++s) {
            const int k0 = ga * GROUPSIZE + s * 32;
            const int r0 = rbase + 4 * s;

            // ---- B source dwords for both N-tiles (issue all loads early)
            const unsigned q00 = (unsigned)qcol0[(size_t)r0 * OUT_F];
            const unsigned q01 = (unsigned)qcol0[(size_t)(r0 + 1) * OUT_F];
            const unsigned q10 = (unsigned)qcol1[(size_t)r0 * OUT_F];
            const unsigned q11 = (unsigned)qcol1[(size_t)(r0 + 1) * OUT_F];
            __builtin_prefetch(qcol0 + (size_t)(r0 + 16) * OUT_F, 0, 3);
            __builtin_prefetch(qcol1 + (size_t)(r0 + 16) * OUT_F, 0, 3);

            // ---- A: preconverted f16, 2x16B per M-tile, feeds 8 WMMAs
            HF A[4];
#pragma unroll
            for (int mt = 0; mt < 4; ++mt) {
                const _Float16* ap = xbase + (size_t)(mt * 16 + l16) * IN_F + k0;
                A[mt].q[0] = *(const h8*)(ap);
                A[mt].q[1] = *(const h8*)(ap + 8);
            }

            HF B0, B1;
#pragma unroll
            for (int p = 0; p < 4; ++p) {
                B0.h[p]     = dq_pair(q00 >> (4 * p), s20, cz0);
                B0.h[4 + p] = dq_pair(q01 >> (4 * p), s20, cz0);
            }
#pragma unroll
            for (int mt = 0; mt < 4; ++mt)
                c[0][mt] = __builtin_amdgcn_wmma_f32_16x16x32_f16(
                               false, A[mt].v, false, B0.v,
                               (short)0, c[0][mt], false, false);
#pragma unroll
            for (int p = 0; p < 4; ++p) {
                B1.h[p]     = dq_pair(q10 >> (4 * p), s21, cz1);
                B1.h[4 + p] = dq_pair(q11 >> (4 * p), s21, cz1);
            }
#pragma unroll
            for (int mt = 0; mt < 4; ++mt)
                c[1][mt] = __builtin_amdgcn_wmma_f32_16x16x32_f16(
                               false, A[mt].v, false, B1.v,
                               (short)0, c[1][mt], false, false);
        }
    }

    // ---- K-split reduction: seed from wave 0, ds_add_f32 from waves 1..7 ----
    if (ks == 0) {
#pragma unroll
        for (int nt = 0; nt < NT; ++nt)
#pragma unroll
            for (int mt = 0; mt < 4; ++mt) {
                float* d = &acc[nt][mt][lane * 8];
#pragma unroll
                for (int j = 0; j < 8; ++j) d[j] = c[nt][mt][j];
            }
    }
    __syncthreads();
    if (ks != 0) {
#pragma unroll
        for (int nt = 0; nt < NT; ++nt)
#pragma unroll
            for (int mt = 0; mt < 4; ++mt)
#pragma unroll
                for (int j = 0; j < 8; ++j)
                    atomicAdd(&acc[nt][mt][lane * 8 + j], c[nt][mt][j]);
    }
    __syncthreads();

    if (ks == 0) {
#pragma unroll
        for (int nt = 0; nt < NT; ++nt) {
            const int n = (nt == 0) ? n0 : n1;
            const float bv = bias[n];
#pragma unroll
            for (int mt = 0; mt < 4; ++mt) {
                float* op = out + (size_t)(mt * 16 + hi * 8) * OUT_F + n;
#pragma unroll
                for (int j = 0; j < 8; ++j)
                    op[(size_t)j * OUT_F] = acc[nt][mt][lane * 8 + j] + bv;
            }
        }
    }
}

extern "C" void kernel_launch(void* const* d_in, const int* in_sizes, int n_in,
                              void* d_out, int out_size, void* d_ws, size_t ws_size,
                              hipStream_t stream) {
    const float* x       = (const float*)d_in[0];
    const int*   qweight = (const int*)d_in[1];
    const int*   qzeros  = (const int*)d_in[2];
    const float* scales  = (const float*)d_in[3];
    const float* bias    = (const float*)d_in[4];
    float*       out     = (float*)d_out;
    _Float16*    xh      = (_Float16*)d_ws;   // 64*8192*2B = 1 MB scratch

    // Pass 1: convert + permute activations
    cvt_x_kernel<<<dim3((BATCH * IN_F / 8) / 256), dim3(256), 0, stream>>>(x, xh);

    // Pass 2: fused dequant GEMM: 256 blocks x 8 waves (= 8 K-splits)
    dim3 grid(OUT_F / 16 / NT);                 // 256
    w4a16_wmma_kernel<<<grid, dim3(256), 0, stream>>>(xh, qweight, qzeros,
                                                      scales, bias, out);
}
